// GlobalSTAR_43207370998211
// MI455X (gfx1250) — matrix operand order repaired and checked by reading, not compiled, not run on hardware
//
#include <hip/hip_runtime.h>
#include <math.h>

// ---------------------------------------------------------------------------
// GlobalSTAR fused kernel for MI455X (gfx1250), wave32 + WMMA bf16.
//   B=64, T=4096, C=128, D=64.  M = B*T = 262144 tokens.
//   out = x + gelu( [x | core] @ Wf + bf ),  core = gumbel-max pick of
//   cand = gelu(x@W1+b1) @ W2 + b2  (JAX threefry2x32, key (0,42)).
// All weights live in LDS as bf16, pre-transposed to [N][K] so each lane's
// WMMA B-fragment is two contiguous 16-byte runs. Intermediates never touch HBM.
// ---------------------------------------------------------------------------

typedef __attribute__((ext_vector_type(16))) __bf16       bf16x16;
typedef __attribute__((ext_vector_type(8)))  float        f32x8;
typedef __attribute__((ext_vector_type(4)))  unsigned int u32x4;

union Frag { bf16x16 bf; u32x4 q[2]; };

__device__ __forceinline__ unsigned short f2bf(float f) {
  unsigned int u = __float_as_uint(f);
  unsigned int r = u + 0x7FFFu + ((u >> 16) & 1u);   // round-to-nearest-even
  return (unsigned short)(r >> 16);
}

__device__ __forceinline__ float gelu_exact(float v) {
  return 0.5f * v * (1.0f + erff(v * 0.70710678118654752440f));
}

__device__ __forceinline__ unsigned int rotl32(unsigned int x, int r) {
  return (x << r) | (x >> (32 - r));
}

// JAX threefry2x32, 20 rounds, key = PRNGKey(42) = (0, 42)
__device__ __forceinline__ void threefry2x32_42(unsigned int c0, unsigned int c1,
                                                unsigned int& o0, unsigned int& o1) {
  const unsigned int ks0 = 0u, ks1 = 42u;
  const unsigned int ks2 = 0x1BD11BDAu ^ ks0 ^ ks1;
  const int RA[4] = {13, 15, 26, 6};
  const int RB[4] = {17, 29, 16, 24};
  unsigned int x0 = c0 + ks0, x1 = c1 + ks1;
#pragma unroll
  for (int i = 0; i < 4; ++i) { x0 += x1; x1 = rotl32(x1, RA[i]); x1 ^= x0; }
  x0 += ks1; x1 += ks2 + 1u;
#pragma unroll
  for (int i = 0; i < 4; ++i) { x0 += x1; x1 = rotl32(x1, RB[i]); x1 ^= x0; }
  x0 += ks2; x1 += ks0 + 2u;
#pragma unroll
  for (int i = 0; i < 4; ++i) { x0 += x1; x1 = rotl32(x1, RA[i]); x1 ^= x0; }
  x0 += ks0; x1 += ks1 + 3u;
#pragma unroll
  for (int i = 0; i < 4; ++i) { x0 += x1; x1 = rotl32(x1, RB[i]); x1 ^= x0; }
  x0 += ks1; x1 += ks2 + 4u;
#pragma unroll
  for (int i = 0; i < 4; ++i) { x0 += x1; x1 = rotl32(x1, RA[i]); x1 ^= x0; }
  x0 += ks2; x1 += ks0 + 5u;
  o0 = x0; o1 = x1;
}

__device__ __forceinline__ f32x8 wmma_bf16(const Frag& a, const Frag& b, f32x8 c) {
  return __builtin_amdgcn_wmma_f32_16x16x32_bf16(false, a.bf, false, b.bf,
                                                 (short)0, c, false, false);
}

// A fragment (16x32 bf16, M x K), base is row-major [rows][ldk] in LDS.
// Lane layout (ISA 7.12.2): hi = lane>>4 selects K-runs [hi*8,+8) and [hi*8+16,+8).
__device__ __forceinline__ Frag load_a(const unsigned short* base, int ldk,
                                       int m, int kk, int hi) {
  Frag f;
  const unsigned short* p = base + m * ldk + kk + hi * 8;
  f.q[0] = *(const u32x4*)p;
  f.q[1] = *(const u32x4*)(p + 16);
  return f;
}

// B fragment (32x16 bf16, K x N) from a TRANSPOSED weight [N][ldk] in LDS.
// Lane n = lane&15 holds K-runs [kk+hi*8,+8) and [kk+hi*8+16,+8).
__device__ __forceinline__ Frag load_b(const unsigned short* base, int ldk,
                                       int n, int kk, int hi) {
  Frag f;
  const unsigned short* p = base + n * ldk + kk + hi * 8;
  f.q[0] = *(const u32x4*)p;
  f.q[1] = *(const u32x4*)(p + 16);
  return f;
}

// ---------------------------------------------------------------------------
// Kernel 0: convert fp32 weights -> bf16, transposed to [N][K], into d_ws.
//   ws layout (ushort): W1t [128][128] @0, W2t [64][128] @16384, Wft [128][192] @24576
// ---------------------------------------------------------------------------
__global__ void star_prep(const float* __restrict__ W1, const float* __restrict__ W2,
                          const float* __restrict__ Wf, unsigned short* __restrict__ wq) {
  int i = blockIdx.x * 256 + threadIdx.x;
  if (i < 128 * 128) {                       // W1: [K=128][N=128] -> W1t[n][k]
    int n = i >> 7, k = i & 127;
    wq[i] = f2bf(W1[k * 128 + n]);
  }
  int j = i - 128 * 128;
  if (j >= 0 && j < 64 * 128) {              // W2: [K=128][N=64] -> W2t[n][k]
    int n = j >> 7, k = j & 127;
    wq[16384 + j] = f2bf(W2[k * 64 + n]);
  }
  int l = i - (16384 + 8192);
  if (l >= 0 && l < 128 * 192) {             // Wf: [K=192][N=128] -> Wft[n][k]
    int n = l / 192, k = l % 192;
    wq[24576 + l] = f2bf(Wf[k * 128 + n]);
  }
}

// ---------------------------------------------------------------------------
// Main fused kernel: 128 threads = 4 waves, each wave owns a 16-token slab.
// ---------------------------------------------------------------------------
__global__ __launch_bounds__(128) void star_main(
    const float* __restrict__ x, const float* __restrict__ b1,
    const float* __restrict__ b2, const float* __restrict__ bfv,
    const unsigned short* __restrict__ wq, float* __restrict__ out) {

  __shared__ __align__(16) unsigned short s_w1t[128 * 128];   // 32 KB
  __shared__ __align__(16) unsigned short s_w2t[64 * 128];    // 16 KB
  __shared__ __align__(16) unsigned short s_wft[128 * 192];   // 48 KB
  __shared__ __align__(16) unsigned short s_x[64 * 128];      // 16 KB (bf16 x tile)
  __shared__ __align__(16) unsigned short s_h[64 * 128];      // 16 KB
  __shared__ __align__(16) float          s_cand[64 * 64];    // 16 KB
  __shared__ __align__(16) unsigned short s_core[64 * 64];    // 8 KB
  __shared__ float s_rbest[128];
  __shared__ int   s_ridx[128];

  const int tid  = threadIdx.x;
  const int wave = tid >> 5;       // 0..3 : slab id
  const int lane = tid & 31;
  const int hi   = lane >> 4;      // half-wave select
  const int m    = lane & 15;      // N column (B/C/D) or M row (A) within tile
  const long long row0 = (long long)blockIdx.x * 64;   // first token of block

  // Warm the x tile for the epilogue residual re-read (global_prefetch_b8):
  // 128 threads x 2 lines of 128B cover the 32KB fp32 tile.
  {
    const float* px = x + row0 * 128 + tid * 64;
    __builtin_prefetch(px, 0, 3);
    __builtin_prefetch(px + 32, 0, 3);
  }

  // ---- cooperative load: weights (bf16, pre-transposed) + x tile ----------
  {
    u32x4*       dw1 = (u32x4*)s_w1t;
    u32x4*       dw2 = (u32x4*)s_w2t;
    u32x4*       dwf = (u32x4*)s_wft;
    const u32x4* sw1 = (const u32x4*)(wq);
    const u32x4* sw2 = (const u32x4*)(wq + 16384);
    const u32x4* swf = (const u32x4*)(wq + 24576);
    for (int i = tid; i < 2048; i += 128) dw1[i] = sw1[i];
    for (int i = tid; i < 1024; i += 128) dw2[i] = sw2[i];
    for (int i = tid; i < 3072; i += 128) dwf[i] = swf[i];
    const float* xb = x + row0 * 128;
    for (int i = tid; i < 64 * 128; i += 128) s_x[i] = f2bf(xb[i]);
  }
  __syncthreads();

  const unsigned short* xs  = s_x    + wave * 16 * 128;
  unsigned short*       hs  = s_h    + wave * 16 * 128;
  float*                cs  = s_cand + wave * 16 * 64;
  unsigned short*       cexp = s_core + wave * 16 * 64;
  const long long trow0 = row0 + (long long)wave * 16;

  // ---- GEMM1: h = gelu(x @ W1 + b1)   [16x128] ----------------------------
  {
    Frag a[4];
#pragma unroll
    for (int k = 0; k < 4; ++k) a[k] = load_a(xs, 128, m, k * 32, hi);
#pragma unroll
    for (int nt = 0; nt < 8; ++nt) {
      f32x8 acc = {0.f, 0.f, 0.f, 0.f, 0.f, 0.f, 0.f, 0.f};
#pragma unroll
      for (int k = 0; k < 4; ++k) {
        Frag b = load_b(s_w1t, 128, nt * 16 + m, k * 32, hi);
        acc = wmma_bf16(a[k], b, acc);
      }
      float bias = b1[nt * 16 + m];
#pragma unroll
      for (int v = 0; v < 8; ++v) {
        float g = gelu_exact(acc[v] + bias);
        hs[(v + hi * 8) * 128 + nt * 16 + m] = f2bf(g);
      }
    }
  }

  // ---- GEMM2: cand = h @ W2 + b2   [16x64] --------------------------------
  {
    Frag a[4];
#pragma unroll
    for (int k = 0; k < 4; ++k) a[k] = load_a(hs, 128, m, k * 32, hi);
#pragma unroll
    for (int nt = 0; nt < 4; ++nt) {
      f32x8 acc = {0.f, 0.f, 0.f, 0.f, 0.f, 0.f, 0.f, 0.f};
#pragma unroll
      for (int k = 0; k < 4; ++k) {
        Frag b = load_b(s_w2t, 128, nt * 16 + m, k * 32, hi);
        acc = wmma_bf16(a[k], b, acc);
      }
      float bias = b2[nt * 16 + m];
#pragma unroll
      for (int v = 0; v < 8; ++v)
        cs[(v + hi * 8) * 64 + nt * 16 + m] = acc[v] + bias;
    }
  }

  // ---- Gumbel-max categorical (JAX key 42), per token row -----------------
  {
    const int row = m;                 // each row handled by lane pair (hi=0/1)
    const int dbase = hi * 32;
    const unsigned int HALF = 1u << 23;          // N = B*T*D = 2^24
    const unsigned int tokbase = (unsigned int)((trow0 + row) * 64);
    float best = -3.402823466e+38f;
    int   bidx = 0;
    for (int j = 0; j < 32; ++j) {
      int d = dbase + j;
      unsigned int i = tokbase + (unsigned int)d;
      unsigned int o0, o1, bits;
      if (i < HALF) { threefry2x32_42(i, i + HALF, o0, o1); bits = o0; }
      else          { threefry2x32_42(i - HALF, i, o0, o1); bits = o1; }
      float u01 = __uint_as_float((bits >> 9) | 0x3F800000u) - 1.0f;
      const float TINY = 1.17549435e-38f;
      float uu = fmaxf(TINY, u01 * (1.0f - TINY) + TINY);
      float g = -logf(-logf(uu));
      float s = cs[row * 64 + d] + g;
      if (s > best) { best = s; bidx = d; }
    }
    // combine the two halves of each row via LDS exchange (wave-internal)
    s_rbest[tid] = best;
    s_ridx[tid]  = bidx;
    float ob = s_rbest[tid ^ 16];
    int   oi = s_ridx[tid ^ 16];
    if (ob > best || (ob == best && oi < bidx)) { best = ob; bidx = oi; }
    unsigned short cb = f2bf(cs[row * 64 + bidx]);   // broadcast core value
    for (int j = 0; j < 32; ++j) cexp[row * 64 + dbase + j] = cb;
  }

  // ---- GEMM3: out = x + gelu([x | core] @ Wf + bf)   K = 192 --------------
  {
    Frag a[6];
#pragma unroll
    for (int k = 0; k < 4; ++k) a[k] = load_a(xs, 128, m, k * 32, hi);
#pragma unroll
    for (int k = 4; k < 6; ++k) a[k] = load_a(cexp, 64, m, (k - 4) * 32, hi);
#pragma unroll
    for (int nt = 0; nt < 8; ++nt) {
      f32x8 acc = {0.f, 0.f, 0.f, 0.f, 0.f, 0.f, 0.f, 0.f};
#pragma unroll
      for (int k = 0; k < 6; ++k) {
        Frag b = load_b(s_wft, 192, nt * 16 + m, k * 32, hi);
        acc = wmma_bf16(a[k], b, acc);
      }
      float bias = bfv[nt * 16 + m];
#pragma unroll
      for (int v = 0; v < 8; ++v) {
        long long gi = (trow0 + (v + hi * 8)) * 128 + nt * 16 + m;
        out[gi] = x[gi] + gelu_exact(acc[v] + bias);
      }
    }
  }
}

// ---------------------------------------------------------------------------
extern "C" void kernel_launch(void* const* d_in, const int* in_sizes, int n_in,
                              void* d_out, int out_size, void* d_ws, size_t ws_size,
                              hipStream_t stream) {
  (void)in_sizes; (void)n_in; (void)out_size; (void)ws_size;
  const float* x  = (const float*)d_in[0];
  const float* W1 = (const float*)d_in[1];
  const float* b1 = (const float*)d_in[2];
  const float* W2 = (const float*)d_in[3];
  const float* b2 = (const float*)d_in[4];
  const float* Wf = (const float*)d_in[5];
  const float* bf_ = (const float*)d_in[6];
  float* out = (float*)d_out;
  unsigned short* wq = (unsigned short*)d_ws;   // needs 98,304 bytes

  // Kernel 0: weight convert/transpose to bf16 (49152 elements)
  star_prep<<<192, 256, 0, stream>>>(W1, W2, Wf, wq);
  // Main fused kernel: 262144 tokens / 64 per block
  star_main<<<4096, 128, 0, stream>>>(x, b1, b2, bf_, wq, out);
}
